// Search_Network_86577950753149
// MI455X (gfx1250) — matrix operand order repaired and checked by reading, not compiled, not run on hardware
//
#include <hip/hip_runtime.h>
#include <math.h>

// ---------------------------------------------------------------------------
// Types for CDNA5 WMMA (wave32, V_WMMA_F32_16X16X32_BF16)
// ---------------------------------------------------------------------------
typedef __attribute__((ext_vector_type(16))) __bf16          v16bf;
typedef __attribute__((ext_vector_type(8)))  float           v8f;
typedef __attribute__((ext_vector_type(8)))  unsigned short  u16x8;
typedef __attribute__((ext_vector_type(4)))  unsigned int    u32x4;
typedef __attribute__((ext_vector_type(4)))  float           f32x4;

union Frag {
    v16bf b;
    u16x8 h[2];
};

static __device__ inline unsigned short f2bf(float f) {
    // round-to-nearest-even fp32 -> bf16
    unsigned u = __float_as_uint(f);
    u += 0x7FFFu + ((u >> 16) & 1u);
    return (unsigned short)(u >> 16);
}

static __device__ inline v8f wmma_bf16(const Frag& a, const Frag& b, v8f c) {
    return __builtin_amdgcn_wmma_f32_16x16x32_bf16(
        /*neg_a=*/false, a.b, /*neg_b=*/false, b.b,
        /*c_mod=*/(short)0, c, /*reuse_a=*/false, /*reuse_b=*/false);
}

#define STRA 40   // LDS A row stride (elems): 80B -> b128 slices aligned, conflict-free
#define STRB 40   // LDS B row stride (elems)

// sizes
#define TT 4
#define NN 8192
#define HH 256
// bf16 weight pool element offsets
#define OFF_LIN1   0
#define OFF_LIN    65536
#define OFF_RLIN   196608
#define OFF_CONV   720896
#define OFF_RCONV  1114112
#define W16_TOTAL  2686976

// ---------------------------------------------------------------------------
// Kernel 1: convert all GEMM weights to bf16, K-major [d][e].
// conv weights are [Dout][Din][3] in the input -> emit per-tap [d][e] matrices.
// ---------------------------------------------------------------------------
__global__ __launch_bounds__(256) void convert_weights(
    const float* __restrict__ lin1_w, const float* __restrict__ lin_w,
    const float* __restrict__ rel_lin_w, const float* __restrict__ conv_w,
    const float* __restrict__ rel_conv_w, unsigned short* __restrict__ w16)
{
    int i = blockIdx.x * 256 + threadIdx.x;
    if (i >= W16_TOTAL) return;
    float v;
    if (i < OFF_LIN) {
        v = lin1_w[i];
    } else if (i < OFF_RLIN) {
        v = lin_w[i - OFF_LIN];
    } else if (i < OFF_CONV) {
        v = rel_lin_w[i - OFF_RLIN];
    } else if (i < OFF_RCONV) {
        int r = i - OFF_CONV;             // [l][tap][d][e]
        int l = r / 196608; r %= 196608;
        int k = r / 65536;  r %= 65536;
        int d = r / 256;    int e = r % 256;
        v = conv_w[(((size_t)l * 256 + e) * 256 + d) * 3 + k];
    } else {
        int r = i - OFF_RCONV;            // [l*4+t][tap][d][e]
        int lt = r / 196608; r %= 196608;
        int k  = r / 65536;  r %= 65536;
        int d  = r / 256;    int e = r % 256;
        v = rel_conv_w[(((size_t)lt * 256 + e) * 256 + d) * 3 + k];
    }
    w16[i] = f2bf(v);
}

// ---------------------------------------------------------------------------
// Kernel 2: h = x @ lin1_w + lin1_b   (M=32768, N=256, K=256), no relu.
// Block = 8 waves, tile 128(M) x 64(N); wave w -> rows [w*16, w*16+16).
// Software-pipelined global prefetch; all 4 B fragments preloaded into
// distinct registers so the 4 WMMAs issue back-to-back after one wait.
// ---------------------------------------------------------------------------
__global__ __launch_bounds__(256) void gemm_lin1(
    const float* __restrict__ x, const unsigned short* __restrict__ w16,
    const float* __restrict__ bias,
    float* __restrict__ h32, unsigned short* __restrict__ h16)
{
    __shared__ __align__(16) unsigned short lds_a[128 * STRA];
    __shared__ __align__(16) unsigned short lds_b[64 * STRB];

    const int tid  = threadIdx.x;
    const int wave = tid >> 5;
    const int lane = tid & 31;
    const int m0   = blockIdx.x * 128;
    const int e0   = blockIdx.y * 64;
    const int m  = lane & 15;
    const int hi = lane >> 4;

    // staging assignments
    const int sArow  = tid >> 1;             // 0..127
    const int sAhalf = (tid & 1) * 16;       // 0 / 16
    const int sBk    = tid & 31;             // 0..31
    const int sBseg  = tid >> 5;             // 0..7

    v8f acc[4] = {v8f{0.f}, v8f{0.f}, v8f{0.f}, v8f{0.f}};

    // prefetch registers
    f32x4 aR0, aR1, aR2, aR3;
    u16x8 bR;
    {
        const float* src = x + (size_t)(m0 + sArow) * 256 + 0 + sAhalf;
        aR0 = ((const f32x4*)src)[0]; aR1 = ((const f32x4*)src)[1];
        aR2 = ((const f32x4*)src)[2]; aR3 = ((const f32x4*)src)[3];
        bR  = *(const u16x8*)(w16 + (size_t)(0 + sBk) * 256 + e0 + sBseg * 8);
    }

    const int abase = (wave * 16 + m) * STRA + hi * 8;

    for (int kb = 0; kb < 256; kb += 32) {
        __syncthreads();
        {   // store staged A (fp32 -> bf16) and B (transposed [n][k]) to LDS
            __align__(16) unsigned short tmp[16];
            #pragma unroll
            for (int i = 0; i < 4; ++i) tmp[i]      = f2bf(aR0[i]);
            #pragma unroll
            for (int i = 0; i < 4; ++i) tmp[4 + i]  = f2bf(aR1[i]);
            #pragma unroll
            for (int i = 0; i < 4; ++i) tmp[8 + i]  = f2bf(aR2[i]);
            #pragma unroll
            for (int i = 0; i < 4; ++i) tmp[12 + i] = f2bf(aR3[i]);
            u16x8* dst = (u16x8*)&lds_a[sArow * STRA + sAhalf];
            dst[0] = *(const u16x8*)&tmp[0];
            dst[1] = *(const u16x8*)&tmp[8];
            #pragma unroll
            for (int i = 0; i < 8; ++i) lds_b[(sBseg * 8 + i) * STRB + sBk] = bR[i];
        }
        __syncthreads();

        if (kb + 32 < 256) {   // prefetch next K-step during compute
            const float* src = x + (size_t)(m0 + sArow) * 256 + (kb + 32) + sAhalf;
            aR0 = ((const f32x4*)src)[0]; aR1 = ((const f32x4*)src)[1];
            aR2 = ((const f32x4*)src)[2]; aR3 = ((const f32x4*)src)[3];
            bR  = *(const u16x8*)(w16 + (size_t)(kb + 32 + sBk) * 256 + e0 + sBseg * 8);
        }

        Frag a;
        a.h[0] = *(const u16x8*)&lds_a[abase];
        a.h[1] = *(const u16x8*)&lds_a[abase + 16];

        // all 4 B fragments in distinct registers, then 4 back-to-back WMMAs
        Frag bg[4];
        #pragma unroll
        for (int s = 0; s < 4; ++s) {
            bg[s].h[0] = *(const u16x8*)&lds_b[(s * 16 + m) * STRB + hi * 8];
            bg[s].h[1] = *(const u16x8*)&lds_b[(s * 16 + m) * STRB + hi * 8 + 16];
        }
        acc[0] = wmma_bf16(a, bg[0], acc[0]);
        acc[1] = wmma_bf16(a, bg[1], acc[1]);
        acc[2] = wmma_bf16(a, bg[2], acc[2]);
        acc[3] = wmma_bf16(a, bg[3], acc[3]);
    }

    #pragma unroll
    for (int s = 0; s < 4; ++s) {
        int e = e0 + s * 16 + m;
        float bv = bias[e];
        #pragma unroll
        for (int j = 0; j < 8; ++j) {
            size_t row = (size_t)(m0 + wave * 16 + hi * 8 + j);
            float v = acc[s][j] + bv;
            h32[row * 256 + e] = v;
            h16[row * 256 + e] = f2bf(v);
        }
    }
}

// ---------------------------------------------------------------------------
// Kernel 3: fused layer. For each tile computes all 4 relu options
// (lin / rel-lin / conv(3 taps) / rel-conv(3 taps)) with shared A staging,
// then mixes with softmax(trans_arch[l]) + identity.
// Grid: (8192/64, 256/64, T). Block = 8 waves arranged 4(M) x 2(N);
// wave tile = 16(M) x 32(N) -> 8 f32 accumulators, 16 WMMAs per K-step.
// B fragments staged in two ping-pong groups of 4 distinct registers:
// each group's 4 WMMAs overlap the next group's 8 in-flight ds_load_b128.
// ---------------------------------------------------------------------------
__global__ __launch_bounds__(256) void layer_fused(
    const unsigned short* __restrict__ w16,
    const float* __restrict__ lin_b, const float* __restrict__ rel_lin_b,
    const float* __restrict__ conv_b, const float* __restrict__ rel_conv_b,
    const float* __restrict__ trans_arch,
    const float* __restrict__ h32_in, const unsigned short* __restrict__ h16_in,
    float* __restrict__ h32_out, unsigned short* __restrict__ h16_out,
    int l)
{
    __shared__ __align__(16) unsigned short lds_a[66 * STRA];        // rows n0-1 .. n0+64
    __shared__ __align__(16) unsigned short lds_b[8 * 64 * STRB];    // 8 weight tiles

    const int tid  = threadIdx.x;
    const int wave = tid >> 5;
    const int lane = tid & 31;
    const int mw = wave & 3, nw = wave >> 2;
    const int n0 = blockIdx.x * 64;
    const int e0 = blockIdx.y * 64;
    const int t  = blockIdx.z;
    const int m  = lane & 15;
    const int hi = lane >> 4;

    const unsigned short* W[8];
    W[0] = w16 + OFF_LIN   + (size_t)l * 65536;                       // linear
    W[1] = w16 + OFF_RLIN  + (size_t)(l * 4 + t) * 65536;             // rel-linear
    W[2] = w16 + OFF_CONV  + (size_t)(l * 3 + 0) * 65536;             // conv taps
    W[3] = w16 + OFF_CONV  + (size_t)(l * 3 + 1) * 65536;
    W[4] = w16 + OFF_CONV  + (size_t)(l * 3 + 2) * 65536;
    W[5] = w16 + OFF_RCONV + (size_t)((l * 4 + t) * 3 + 0) * 65536;   // rel-conv taps
    W[6] = w16 + OFF_RCONV + (size_t)((l * 4 + t) * 3 + 1) * 65536;
    W[7] = w16 + OFF_RCONV + (size_t)((l * 4 + t) * 3 + 2) * 65536;

    const unsigned short* hin = h16_in + (size_t)t * NN * 256;

    // staging assignments: A: 132 threads, half row (16 bf16) each
    const int sArow  = tid >> 1;            // 0..65 (valid when tid < 132)
    const int sAhalf = (tid & 1) * 16;      // 0 / 16
    const int sAnode = n0 + sArow - 1;
    const bool sAon  = (tid < 132);
    const bool sAin  = sAon && (sAnode >= 0) && (sAnode < NN);
    const int sBk    = tid & 31;            // 0..31
    const int sBseg  = tid >> 5;            // 0..7

    v8f s_lin[2]   = {v8f{0.f}, v8f{0.f}};
    v8f s_rlin[2]  = {v8f{0.f}, v8f{0.f}};
    v8f s_conv[2]  = {v8f{0.f}, v8f{0.f}};
    v8f s_rconv[2] = {v8f{0.f}, v8f{0.f}};

    // prefetch registers
    u32x4 aP0 = {0u,0u,0u,0u}, aP1 = {0u,0u,0u,0u};
    u16x8 bP[8];
    {
        if (sAin) {
            const u32x4* src = (const u32x4*)(hin + (size_t)sAnode * 256 + 0 + sAhalf);
            aP0 = src[0]; aP1 = src[1];
        }
        #pragma unroll
        for (int mtx = 0; mtx < 8; ++mtx)
            bP[mtx] = *(const u16x8*)(W[mtx] + (size_t)(0 + sBk) * 256 + e0 + sBseg * 8);
    }

    const int arow = 1 + mw * 16 + m;

    for (int kb = 0; kb < 256; kb += 32) {
        __syncthreads();
        {   // store staged tiles to LDS
            if (sAon) {
                u16x8* dst = (u16x8*)&lds_a[sArow * STRA + sAhalf];
                union { u32x4 u; u16x8 s; } c0, c1;
                c0.u = aP0; c1.u = aP1;
                dst[0] = c0.s; dst[1] = c1.s;
            }
            #pragma unroll
            for (int mtx = 0; mtx < 8; ++mtx) {
                unsigned short* bb = &lds_b[mtx * 64 * STRB];
                #pragma unroll
                for (int i = 0; i < 8; ++i) bb[(sBseg * 8 + i) * STRB + sBk] = bP[mtx][i];
            }
        }
        __syncthreads();

        if (kb + 32 < 256) {   // prefetch next K-step during compute
            if (sAin) {
                const u32x4* src = (const u32x4*)(hin + (size_t)sAnode * 256 + (kb + 32) + sAhalf);
                aP0 = src[0]; aP1 = src[1];
            }
            #pragma unroll
            for (int mtx = 0; mtx < 8; ++mtx)
                bP[mtx] = *(const u16x8*)(W[mtx] + (size_t)(kb + 32 + sBk) * 256 + e0 + sBseg * 8);
        }

        Frag am, a0, ap;
        am.h[0] = *(const u16x8*)&lds_a[(arow - 1) * STRA + hi * 8];
        am.h[1] = *(const u16x8*)&lds_a[(arow - 1) * STRA + hi * 8 + 16];
        a0.h[0] = *(const u16x8*)&lds_a[arow * STRA + hi * 8];
        a0.h[1] = *(const u16x8*)&lds_a[arow * STRA + hi * 8 + 16];
        ap.h[0] = *(const u16x8*)&lds_a[(arow + 1) * STRA + hi * 8];
        ap.h[1] = *(const u16x8*)&lds_a[(arow + 1) * STRA + hi * 8 + 16];

        const int nc0 = (nw * 32 + m) * STRB + hi * 8;
        const int nc1 = (nw * 32 + 16 + m) * STRB + hi * 8;

        #define LOADB(fr, mtx, nc) \
            fr.h[0] = *(const u16x8*)&lds_b[(mtx) * 64 * STRB + (nc)]; \
            fr.h[1] = *(const u16x8*)&lds_b[(mtx) * 64 * STRB + (nc) + 16];

        // Two ping-pong groups of 4 fragments in DISTINCT registers.
        Frag g0a, g0b, g0c, g0d;   // group A
        Frag g1a, g1b, g1c, g1d;   // group B

        // subtile 0, matrices {0,2,1,3}
        LOADB(g0a, 0, nc0); LOADB(g0b, 2, nc0); LOADB(g0c, 1, nc0); LOADB(g0d, 3, nc0);
        // subtile 0, matrices {5,4,6,7}
        LOADB(g1a, 5, nc0); LOADB(g1b, 4, nc0); LOADB(g1c, 6, nc0); LOADB(g1d, 7, nc0);

        s_lin[0]   = wmma_bf16(a0, g0a, s_lin[0]);
        s_conv[0]  = wmma_bf16(am, g0b, s_conv[0]);
        s_rlin[0]  = wmma_bf16(a0, g0c, s_rlin[0]);
        s_conv[0]  = wmma_bf16(a0, g0d, s_conv[0]);

        // subtile 1, matrices {0,2,1,3} (overlaps group B's WMMAs)
        LOADB(g0a, 0, nc1); LOADB(g0b, 2, nc1); LOADB(g0c, 1, nc1); LOADB(g0d, 3, nc1);

        s_rconv[0] = wmma_bf16(am, g1a, s_rconv[0]);
        s_conv[0]  = wmma_bf16(ap, g1b, s_conv[0]);
        s_rconv[0] = wmma_bf16(a0, g1c, s_rconv[0]);
        s_rconv[0] = wmma_bf16(ap, g1d, s_rconv[0]);

        // subtile 1, matrices {5,4,6,7}
        LOADB(g1a, 5, nc1); LOADB(g1b, 4, nc1); LOADB(g1c, 6, nc1); LOADB(g1d, 7, nc1);

        s_lin[1]   = wmma_bf16(a0, g0a, s_lin[1]);
        s_conv[1]  = wmma_bf16(am, g0b, s_conv[1]);
        s_rlin[1]  = wmma_bf16(a0, g0c, s_rlin[1]);
        s_conv[1]  = wmma_bf16(a0, g0d, s_conv[1]);

        s_rconv[1] = wmma_bf16(am, g1a, s_rconv[1]);
        s_conv[1]  = wmma_bf16(ap, g1b, s_conv[1]);
        s_rconv[1] = wmma_bf16(a0, g1c, s_rconv[1]);
        s_rconv[1] = wmma_bf16(ap, g1d, s_rconv[1]);
        #undef LOADB
    }

    // softmax(trans_arch[l] / TEMP), TEMP = 1
    float ta[5], tmax = -3.4e38f;
    #pragma unroll
    for (int i = 0; i < 5; ++i) { ta[i] = trans_arch[l * 5 + i]; tmax = fmaxf(tmax, ta[i]); }
    float tsum = 0.f;
    #pragma unroll
    for (int i = 0; i < 5; ++i) { ta[i] = __expf(ta[i] - tmax); tsum += ta[i]; }
    const float inv = 1.f / tsum;
    const float tw0 = ta[0]*inv, tw1 = ta[1]*inv, tw2 = ta[2]*inv, tw3 = ta[3]*inv, tw4 = ta[4]*inv;

    #pragma unroll
    for (int s = 0; s < 2; ++s) {
        const int e = e0 + nw * 32 + s * 16 + m;
        const float blin   = lin_b[l * 256 + e];
        const float brlin  = rel_lin_b[(l * 4 + t) * 256 + e];
        const float bconv  = conv_b[l * 256 + e];
        const float brconv = rel_conv_b[(l * 4 + t) * 256 + e];
        #pragma unroll
        for (int j = 0; j < 8; ++j) {
            size_t row = (size_t)t * NN + n0 + mw * 16 + hi * 8 + j;
            float hold = h32_in[row * 256 + e];
            float v = tw0 * fmaxf(s_lin[s][j]   + blin,   0.f)
                    + tw1 * fmaxf(s_rlin[s][j]  + brlin,  0.f)
                    + tw2 * fmaxf(s_conv[s][j]  + bconv,  0.f)
                    + tw3 * fmaxf(s_rconv[s][j] + brconv, 0.f)
                    + tw4 * hold;
            h32_out[row * 256 + e] = v;
            h16_out[row * 256 + e] = f2bf(v);
        }
    }
}

// ---------------------------------------------------------------------------
// Kernel 4: per-block partial column sums / maxes over 32768 rows.
// ---------------------------------------------------------------------------
__global__ __launch_bounds__(256) void reduce_partial(
    const float* __restrict__ h32, float* __restrict__ psum, float* __restrict__ pmax)
{
    const int e = threadIdx.x;
    const int b = blockIdx.x;
    float s = 0.f, mx = -3.4e38f;
    const float* base = h32 + (size_t)b * 256 * 256 + e;
    for (int r = 0; r < 256; ++r) {
        float v = base[(size_t)r * 256];
        s += v;
        mx = fmaxf(mx, v);
    }
    psum[b * 256 + e] = s;
    pmax[b * 256 + e] = mx;
}

// ---------------------------------------------------------------------------
// Kernel 5: final reduce + softmax(agg_arch) mix + [256]x[256,64] output head.
// ---------------------------------------------------------------------------
__global__ __launch_bounds__(256) void final_head(
    const float* __restrict__ psum, const float* __restrict__ pmax,
    const float* __restrict__ agg_arch, const float* __restrict__ out_w,
    const float* __restrict__ out_b, float* __restrict__ out)
{
    __shared__ float agg[256];
    const int e = threadIdx.x;
    float s = 0.f, mx = -3.4e38f;
    for (int b = 0; b < 128; ++b) {
        s += psum[b * 256 + e];
        mx = fmaxf(mx, pmax[b * 256 + e]);
    }
    float a0 = agg_arch[0], a1 = agg_arch[1], a2 = agg_arch[2];
    float am = fmaxf(a0, fmaxf(a1, a2));
    float e0 = __expf(a0 - am), e1 = __expf(a1 - am), e2 = __expf(a2 - am);
    float inv = 1.f / (e0 + e1 + e2);
    agg[e] = (e0 * s + e1 * (s * (1.f / 32768.f)) + e2 * mx) * inv;
    __syncthreads();
    if (e < 64) {
        float acc = out_b[e];
        for (int k = 0; k < 256; ++k) acc += agg[k] * out_w[k * 64 + e];
        out[e] = acc;
    }
}

// ---------------------------------------------------------------------------
// Host launcher
// ---------------------------------------------------------------------------
extern "C" void kernel_launch(void* const* d_in, const int* in_sizes, int n_in,
                              void* d_out, int out_size, void* d_ws, size_t ws_size,
                              hipStream_t stream)
{
    const float* x          = (const float*)d_in[0];
    const float* lin1_w     = (const float*)d_in[1];
    const float* lin1_b     = (const float*)d_in[2];
    const float* lin_w      = (const float*)d_in[3];
    const float* lin_b      = (const float*)d_in[4];
    const float* rel_lin_w  = (const float*)d_in[5];
    const float* rel_lin_b  = (const float*)d_in[6];
    const float* conv_w     = (const float*)d_in[7];
    const float* conv_b     = (const float*)d_in[8];
    const float* rel_conv_w = (const float*)d_in[9];
    const float* rel_conv_b = (const float*)d_in[10];
    const float* out_w      = (const float*)d_in[11];
    const float* out_b      = (const float*)d_in[12];
    const float* trans_arch = (const float*)d_in[13];
    const float* agg_arch   = (const float*)d_in[14];

    char* ws = (char*)d_ws;
    unsigned short* w16  = (unsigned short*)ws;                 // 5,373,952 B
    float*          h32a = (float*)(ws + 0x0520000);            // 32 MB
    float*          h32b = (float*)(ws + 0x2520000);            // 32 MB
    unsigned short* h16a = (unsigned short*)(ws + 0x4520000);   // 16 MB
    unsigned short* h16b = (unsigned short*)(ws + 0x5520000);   // 16 MB
    float*          psum = (float*)(ws + 0x6520000);            // 128*256*4
    float*          pmax = (float*)(ws + 0x6540000);

    convert_weights<<<(W16_TOTAL + 255) / 256, 256, 0, stream>>>(
        lin1_w, lin_w, rel_lin_w, conv_w, rel_conv_w, w16);

    gemm_lin1<<<dim3(256, 4), 256, 0, stream>>>(x, w16 + OFF_LIN1, lin1_b, h32a, h16a);

    layer_fused<<<dim3(128, 4, TT), 256, 0, stream>>>(
        w16, lin_b, rel_lin_b, conv_b, rel_conv_b, trans_arch,
        h32a, h16a, h32b, h16b, 0);
    layer_fused<<<dim3(128, 4, TT), 256, 0, stream>>>(
        w16, lin_b, rel_lin_b, conv_b, rel_conv_b, trans_arch,
        h32b, h16b, h32a, h16a, 1);

    reduce_partial<<<128, 256, 0, stream>>>(h32a, psum, pmax);
    final_head<<<1, 256, 0, stream>>>(psum, pmax, agg_arch, out_w, out_b, (float*)d_out);
}
